// trajectory_aware_attention_28063316312857
// MI455X (gfx1250) — matrix operand order repaired and checked by reading, not compiled, not run on hardware
//
#include <hip/hip_runtime.h>
#include <math.h>

// ---------------------------------------------------------------------------
// Types for CDNA5 WMMA (wave32, 16x16x32 bf16 -> f32)
// ---------------------------------------------------------------------------
typedef __bf16 bf16;
typedef __attribute__((ext_vector_type(16))) __bf16 v16bf;
typedef __attribute__((ext_vector_type(8)))  float  v8f;

#define SCALE_F 0.08838834764831845f   // 128^-0.5

static __device__ __forceinline__ bf16 f2bf(float f) {
  union { float f; unsigned u; } v; v.f = f;
  unsigned r = v.u + 0x7FFFu + ((v.u >> 16) & 1u);   // round-to-nearest-even
  unsigned short h = (unsigned short)(r >> 16);
  return __builtin_bit_cast(bf16, h);
}

static __device__ __forceinline__ v8f wmma_bf16(v16bf a, v16bf b, v8f c) {
  // 8 args: (neg_a, A, neg_b, B, c_mod, C, reuse_a, reuse_b)
  return __builtin_amdgcn_wmma_f32_16x16x32_bf16(false, a, false, b, (short)0, c,
                                                 false, false);
}

// A fragment: 16x32 bf16 from row-major [M x Kdim] tile (K contiguous).
// lane M = mbase + lane%16; lanes 0-15 hold K {0..15}; 16-31 {16..31}.
// -> two contiguous 16B runs per lane => 2x ds_load_b128.
static __device__ __forceinline__ v16bf load_a_frag(const bf16* p, int mbase,
                                                    int kbase, int ld) {
  int lane = threadIdx.x & 31;
  int m = mbase + (lane & 15);
  int g = (lane >> 4) << 3;   // 0 or 8
  v16bf a;
#pragma unroll
  for (int h = 0; h < 2; ++h)
#pragma unroll
    for (int e = 0; e < 8; ++e)
      a[h * 8 + e] = p[(size_t)m * ld + kbase + (g << 1) + h * 8 + e];
  return a;
}

// B fragment (32x16) from a row-major [N x Kdim] tile (K contiguous):
// B[k][n] = tile[nbase+n][kbase+k].  lane N = nbase + lane%16,
// K = 16*(lane/16) + e  -> one contiguous 32B run per lane => 2x ds_load_b128.
static __device__ __forceinline__ v16bf load_b_fragT(const bf16* p, int nbase,
                                                     int kbase, int ld) {
  int lane = threadIdx.x & 31;
  int n = nbase + (lane & 15);
  int kb = kbase + ((lane >> 4) << 4);
  v16bf b;
#pragma unroll
  for (int e = 0; e < 16; ++e)
    b[e] = p[(size_t)n * ld + kb + e];
  return b;
}

// Row reductions across the 16-lane N-group of the C layout (wave32:
// xor masks < 16 stay inside each half).
static __device__ __forceinline__ float rmax16(float x) {
#pragma unroll
  for (int m = 1; m < 16; m <<= 1) x = fmaxf(x, __shfl_xor(x, m, 32));
  return x;
}
static __device__ __forceinline__ float rsum16(float x) {
#pragma unroll
  for (int m = 1; m < 16; m <<= 1) x += __shfl_xor(x, m, 32);
  return x;
}

// ---------------------------------------------------------------------------
// f32 -> bf16 conversion (bandwidth bound)
// ---------------------------------------------------------------------------
__global__ void cvt_f32_bf16(const float* __restrict__ in, bf16* __restrict__ out,
                             long n) {
  long i = (long)blockIdx.x * blockDim.x + threadIdx.x;
  long stride = (long)gridDim.x * blockDim.x;
  for (; i < n; i += stride) out[i] = f2bf(in[i]);
}

// f32 [R x C] -> bf16 transposed [C x R]  (LDS-tiled, both sides coalesced)
__global__ __launch_bounds__(256) void cvt_transpose_bf16(
    const float* __restrict__ in, bf16* __restrict__ out, int R, int C) {
  __shared__ float t[32][33];
  const int tx = threadIdx.x & 31, ty = threadIdx.x >> 5;
  const int c0 = blockIdx.x * 32, r0 = blockIdx.y * 32;
#pragma unroll
  for (int i = 0; i < 32; i += 8)
    t[ty + i][tx] = in[(size_t)(r0 + ty + i) * C + c0 + tx];
  __syncthreads();
#pragma unroll
  for (int i = 0; i < 32; i += 8)
    out[(size_t)(c0 + ty + i) * R + r0 + tx] = f2bf(t[tx][ty + i]);
}

// bf16 V slice of qkv -> Vt[(b*8+h)*128 + d][j]   (LDS-tiled transpose)
__global__ __launch_bounds__(256) void transpose_v_bf16(
    const bf16* __restrict__ qkv, bf16* __restrict__ vt) {
  __shared__ bf16 t[32][33];
  const int tx = threadIdx.x & 31, ty = threadIdx.x >> 5;
  const int bh = blockIdx.z, b = bh >> 3, h = bh & 7;
  const int j0 = blockIdx.x * 32, d0 = blockIdx.y * 32;
#pragma unroll
  for (int i = 0; i < 32; i += 8)
    t[ty + i][tx] =
        qkv[(size_t)(b * 1024 + j0 + ty + i) * 3072 + 2048 + h * 128 + d0 + tx];
  __syncthreads();
#pragma unroll
  for (int i = 0; i < 32; i += 8)
    vt[((size_t)bh * 128 + d0 + ty + i) * 1024 + j0 + tx] = t[tx][ty + i];
}

// ---------------------------------------------------------------------------
// bf16 WMMA GEMM with pre-transposed B:  C[M,N] = A[M,K] * Bt[N,K]^T (+bias).
// M,N % 128 == 0, K % 32 == 0.  256 threads = 8 waves, 128x128 C tile,
// K-step 32, double-buffered LDS with register-staged global loads so the
// next tile's global traffic overlaps the current tile's WMMAs.
// ---------------------------------------------------------------------------
__global__ __launch_bounds__(256) void gemm_bf16_wmma(
    const bf16* __restrict__ A, const bf16* __restrict__ Bt,
    bf16* __restrict__ Cb, float* __restrict__ Cf,
    const float* __restrict__ bias, int M, int N, int K, int store_f32) {
  __shared__ bf16 As[2][128 * 32];
  __shared__ bf16 Bs[2][128 * 32];
  const int tid = threadIdx.x;
  const int lane = tid & 31;
  const int wave = tid >> 5;
  const int row0 = blockIdx.y * 128;
  const int col0 = blockIdx.x * 128;

  const int r = tid >> 1;            // 0..127
  const int c = (tid & 1) * 16;      // 0 or 16
  const bf16* ag = A + (size_t)(row0 + r) * K + c;
  const bf16* bg = Bt + (size_t)(col0 + r) * K + c;

  v8f acc[8] = {};

  // prologue: stage tile 0
  {
    const uint4* sa = reinterpret_cast<const uint4*>(ag);
    const uint4* sb = reinterpret_cast<const uint4*>(bg);
    uint4* da = reinterpret_cast<uint4*>(&As[0][r * 32 + c]);
    uint4* db = reinterpret_cast<uint4*>(&Bs[0][r * 32 + c]);
    da[0] = sa[0]; da[1] = sa[1];
    db[0] = sb[0]; db[1] = sb[1];
  }
  __syncthreads();

  const int nk = K / 32;
  int buf = 0;
  for (int t = 0; t < nk; ++t) {
    uint4 ra0, ra1, rb0, rb1;
    const bool more = (t + 1 < nk);
    if (more) {  // issue next tile's global loads before compute
      const uint4* sa = reinterpret_cast<const uint4*>(ag + (t + 1) * 32);
      const uint4* sb = reinterpret_cast<const uint4*>(bg + (t + 1) * 32);
      ra0 = sa[0]; ra1 = sa[1]; rb0 = sb[0]; rb1 = sb[1];
    }

    v16bf a = load_a_frag(As[buf], wave * 16, 0, 32);
    v16bf bcur = load_b_fragT(Bs[buf], 0, 0, 32);
#pragma unroll
    for (int nn = 0; nn < 8; ++nn) {
      v16bf bnext = bcur;
      if (nn < 7) bnext = load_b_fragT(Bs[buf], (nn + 1) * 16, 0, 32);
      acc[nn] = wmma_bf16(a, bcur, acc[nn]);
      bcur = bnext;
    }

    if (more) {  // stage next tile into the other buffer
      int nb = buf ^ 1;
      uint4* da = reinterpret_cast<uint4*>(&As[nb][r * 32 + c]);
      uint4* db = reinterpret_cast<uint4*>(&Bs[nb][r * 32 + c]);
      da[0] = ra0; da[1] = ra1;
      db[0] = rb0; db[1] = rb1;
    }
    __syncthreads();
    buf ^= 1;
  }

  const int rgrp = (lane >> 4) * 8;
  const int ncol = lane & 15;
#pragma unroll
  for (int nn = 0; nn < 8; ++nn) {
    int colg = col0 + nn * 16 + ncol;
    float bv = store_f32 ? bias[colg] : 0.f;
#pragma unroll
    for (int rr = 0; rr < 8; ++rr) {
      int rowg = row0 + wave * 16 + rgrp + rr;
      if (store_f32)
        Cf[(size_t)rowg * N + colg] = acc[nn][rr] + bv;
      else
        Cb[(size_t)rowg * N + colg] = f2bf(acc[nn][rr]);
    }
  }
}

// ---------------------------------------------------------------------------
// Pass 1: row max / sum(exp) of S_m = q_m * k_m^T  (softmax-m statistics).
// One block = 128 query rows of one (b,h); 8 waves * 16 rows.
// qk layout: [8192, 2048]; q_m cols = h*128, k_m cols = 1024 + h*128.
// ---------------------------------------------------------------------------
__global__ __launch_bounds__(256) void attn_stats(const bf16* __restrict__ qk,
                                                  float* __restrict__ mstat,
                                                  float* __restrict__ lstat) {
  __shared__ bf16 Ks[128 * 128];
  const int tid = threadIdx.x, lane = tid & 31, wave = tid >> 5;
  const int bid = blockIdx.x;
  const int it = bid & 7, h = (bid >> 3) & 7, b = bid >> 6;
  const int NQK = 2048;
  const size_t rowbase = (size_t)b * 1024;
  const int i0 = it * 128;

  {  // stage q_m tile (128x128)
    int r = tid >> 1, c = (tid & 1) * 64;
    const uint4* src = reinterpret_cast<const uint4*>(
        qk + (rowbase + i0 + r) * NQK + h * 128 + c);
    uint4* dst = reinterpret_cast<uint4*>(Ks + r * 128 + c);
#pragma unroll
    for (int x = 0; x < 8; ++x) dst[x] = src[x];
  }
  __syncthreads();
  v16bf aq[4];
#pragma unroll
  for (int kk = 0; kk < 4; ++kk) aq[kk] = load_a_frag(Ks, wave * 16, kk * 32, 128);
  __syncthreads();

  float m[8], l[8];
#pragma unroll
  for (int r = 0; r < 8; ++r) { m[r] = -__builtin_inff(); l[r] = 0.f; }

  for (int jc = 0; jc < 8; ++jc) {
    {  // stage k_m chunk (128x128)
      int r = tid >> 1, c = (tid & 1) * 64;
      const uint4* src = reinterpret_cast<const uint4*>(
          qk + (rowbase + jc * 128 + r) * NQK + 1024 + h * 128 + c);
      uint4* dst = reinterpret_cast<uint4*>(Ks + r * 128 + c);
#pragma unroll
      for (int x = 0; x < 8; ++x) dst[x] = src[x];
    }
    __syncthreads();
#pragma unroll
    for (int jj = 0; jj < 8; ++jj) {
      // batch all four B fragments so DS loads overlap the WMMA chain
      v16bf b0 = load_b_fragT(Ks, jj * 16, 0, 128);
      v16bf b1 = load_b_fragT(Ks, jj * 16, 32, 128);
      v16bf b2 = load_b_fragT(Ks, jj * 16, 64, 128);
      v16bf b3 = load_b_fragT(Ks, jj * 16, 96, 128);
      v8f c = {};
      c = wmma_bf16(aq[0], b0, c);
      c = wmma_bf16(aq[1], b1, c);
      c = wmma_bf16(aq[2], b2, c);
      c = wmma_bf16(aq[3], b3, c);
#pragma unroll
      for (int r = 0; r < 8; ++r) {
        float t = rmax16(c[r]);
        float mn = fmaxf(m[r], t);
        float s = rsum16(__expf(c[r] - mn));
        l[r] = l[r] * __expf(m[r] - mn) + s;
        m[r] = mn;
      }
    }
    __syncthreads();
  }

  if ((lane & 15) == 0) {
    size_t base = ((size_t)(b * 8 + h)) * 1024 + i0 + wave * 16 + (lane >> 4) * 8;
#pragma unroll
    for (int r = 0; r < 8; ++r) { mstat[base + r] = m[r]; lstat[base + r] = l[r]; }
  }
}

// ---------------------------------------------------------------------------
// Pass 2: flash pass over keys. dots = softmax_m(S_m)*S_v*SCALE + g_l, online
// softmax of dots, O = attn @ V.  V is pre-transposed (vt[(bh*128+d)][j]) so
// all B fragments are K-contiguous.  Writes O bf16 [b*1024+n, h*128+d].
// ---------------------------------------------------------------------------
__global__ __launch_bounds__(256) void attn_flash(
    const bf16* __restrict__ qkv, const bf16* __restrict__ qk,
    const bf16* __restrict__ vt, const float* __restrict__ g_l,
    const float* __restrict__ mstat, const float* __restrict__ lstat,
    bf16* __restrict__ O) {
  __shared__ bf16 Km[64 * 128];    // k_m chunk [j][d]
  __shared__ bf16 Kv[64 * 128];    // k_v chunk [j][d]
  __shared__ bf16 Vt[128 * 64];    // v chunk   [d][j]
  __shared__ bf16 Ps[8][16 * 32];  // per-wave P repack scratch

  const int tid = threadIdx.x, lane = tid & 31, wave = tid >> 5;
  const int bid = blockIdx.x;
  const int it = bid & 7, h = (bid >> 3) & 7, b = bid >> 6;
  const int NQK = 2048, NQKV = 3072;
  const size_t rowbase = (size_t)b * 1024;
  const int i0 = it * 128;
  const int bh = b * 8 + h;

  // ---- stage + gather q_m / q_v fragments (reuse Km/Kv for staging) ----
  v16bf aqm[4], aqv[4];
  {
    int r = tid >> 1, c = (tid & 1) * 64;
    const uint4* src = reinterpret_cast<const uint4*>(
        qk + (rowbase + i0 + r) * NQK + h * 128 + c);
    uint4* dst = reinterpret_cast<uint4*>(
        (r < 64 ? Km + r * 128 : Kv + (r - 64) * 128) + c);
#pragma unroll
    for (int x = 0; x < 8; ++x) dst[x] = src[x];
  }
  __syncthreads();
  {
    const bf16* qsrc = (wave < 4) ? Km : Kv;
    int qrow = (wave & 3) * 16;
#pragma unroll
    for (int kk = 0; kk < 4; ++kk) aqm[kk] = load_a_frag(qsrc, qrow, kk * 32, 128);
  }
  __syncthreads();
  {
    int r = tid >> 1, c = (tid & 1) * 64;
    const uint4* src = reinterpret_cast<const uint4*>(
        qkv + (rowbase + i0 + r) * NQKV + h * 128 + c);
    uint4* dst = reinterpret_cast<uint4*>(
        (r < 64 ? Km + r * 128 : Kv + (r - 64) * 128) + c);
#pragma unroll
    for (int x = 0; x < 8; ++x) dst[x] = src[x];
  }
  __syncthreads();
  {
    const bf16* qsrc = (wave < 4) ? Km : Kv;
    int qrow = (wave & 3) * 16;
#pragma unroll
    for (int kk = 0; kk < 4; ++kk) aqv[kk] = load_a_frag(qsrc, qrow, kk * 32, 128);
  }
  __syncthreads();

  // ---- per-row state ----
  float m1[8], l1inv[8], m2[8], l2[8];
  {
    size_t sb = ((size_t)bh) * 1024 + i0 + wave * 16 + (lane >> 4) * 8;
#pragma unroll
    for (int r = 0; r < 8; ++r) {
      m1[r] = mstat[sb + r];
      l1inv[r] = 1.f / lstat[sb + r];
      m2[r] = -__builtin_inff();
      l2[r] = 0.f;
    }
  }
  v8f acc[8] = {};

  // ---- key loop: 16 chunks of 64 keys ----
  for (int jc = 0; jc < 16; ++jc) {
    const int j0 = jc * 64;
    {  // stage k_m / k_v [64 x 128]
      int r = tid >> 2, c = (tid & 3) * 32;
      size_t grow = rowbase + j0 + r;
      const uint4* s1 =
          reinterpret_cast<const uint4*>(qk + grow * NQK + 1024 + h * 128 + c);
      const uint4* s2 =
          reinterpret_cast<const uint4*>(qkv + grow * NQKV + 1024 + h * 128 + c);
      uint4* d1 = reinterpret_cast<uint4*>(Km + r * 128 + c);
      uint4* d2 = reinterpret_cast<uint4*>(Kv + r * 128 + c);
#pragma unroll
      for (int x = 0; x < 4; ++x) { d1[x] = s1[x]; d2[x] = s2[x]; }
    }
    {  // stage Vt [128 d x 64 j]  (rows contiguous in j)
      int d = tid >> 1, c = (tid & 1) * 32;
      const uint4* s3 = reinterpret_cast<const uint4*>(
          vt + ((size_t)bh * 128 + d) * 1024 + j0 + c);
      uint4* d3 = reinterpret_cast<uint4*>(Vt + d * 64 + c);
#pragma unroll
      for (int x = 0; x < 4; ++x) d3[x] = s3[x];
    }
    // prefetch the next g_l chunk into GL2 (dominant HBM stream, 256 MB)
    if (jc + 1 < 16) {
      int irow = i0 + wave * 16 + (lane >> 4) * 8;
      const float* pf =
          &g_l[(((size_t)bh) * 1024 + irow) * 1024 + j0 + 64 + (lane & 15) * 4];
      __builtin_prefetch(pf, 0, 1);
    }
    __syncthreads();

#pragma unroll
    for (int jj2 = 0; jj2 < 2; ++jj2) {    // 32 keys per iteration
      float dots[2][8];
#pragma unroll
      for (int hf = 0; hf < 2; ++hf) {
        int jcol = jj2 * 32 + hf * 16;
        // batch fragment loads so DS latency overlaps the WMMA chains
        v16bf bm0 = load_b_fragT(Km, jcol, 0, 128);
        v16bf bm1 = load_b_fragT(Km, jcol, 32, 128);
        v16bf bm2 = load_b_fragT(Km, jcol, 64, 128);
        v16bf bm3 = load_b_fragT(Km, jcol, 96, 128);
        v8f sm = {};
        sm = wmma_bf16(aqm[0], bm0, sm);
        sm = wmma_bf16(aqm[1], bm1, sm);
        sm = wmma_bf16(aqm[2], bm2, sm);
        sm = wmma_bf16(aqm[3], bm3, sm);
        v16bf bv0 = load_b_fragT(Kv, jcol, 0, 128);
        v16bf bv1 = load_b_fragT(Kv, jcol, 32, 128);
        v16bf bv2 = load_b_fragT(Kv, jcol, 64, 128);
        v16bf bv3 = load_b_fragT(Kv, jcol, 96, 128);
        v8f sv = {};
        sv = wmma_bf16(aqv[0], bv0, sv);
        sv = wmma_bf16(aqv[1], bv1, sv);
        sv = wmma_bf16(aqv[2], bv2, sv);
        sv = wmma_bf16(aqv[3], bv3, sv);
#pragma unroll
        for (int r = 0; r < 8; ++r) {
          int irow = i0 + wave * 16 + (lane >> 4) * 8 + r;
          size_t gidx =
              (((size_t)bh) * 1024 + irow) * 1024 + j0 + jcol + (lane & 15);
          float pm = __expf(sm[r] - m1[r]) * l1inv[r];
          dots[hf][r] = pm * sv[r] * SCALE_F + g_l[gidx];
        }
      }
      // online softmax update over the 32 columns
#pragma unroll
      for (int r = 0; r < 8; ++r) {
        float t = rmax16(fmaxf(dots[0][r], dots[1][r]));
        float mn = fmaxf(m2[r], t);
        float sc = __expf(m2[r] - mn);
        float e0 = __expf(dots[0][r] - mn);
        float e1 = __expf(dots[1][r] - mn);
        l2[r] = l2[r] * sc + rsum16(e0 + e1);
        m2[r] = mn;
#pragma unroll
        for (int nn = 0; nn < 8; ++nn) acc[nn][r] *= sc;
        int prow = (lane >> 4) * 8 + r;
        Ps[wave][prow * 32 + (lane & 15)] = f2bf(e0);
        Ps[wave][prow * 32 + 16 + (lane & 15)] = f2bf(e1);
      }
      // O += P(16x32) @ V(32x128): B[k][n] = Vt[n][jj2*32+k] (K-contiguous)
      v16bf pa = load_a_frag(Ps[wave], 0, 0, 32);
      v16bf bcur = load_b_fragT(Vt, 0, jj2 * 32, 64);
#pragma unroll
      for (int nn = 0; nn < 8; ++nn) {
        v16bf bnext = bcur;
        if (nn < 7) bnext = load_b_fragT(Vt, (nn + 1) * 16, jj2 * 32, 64);
        acc[nn] = wmma_bf16(pa, bcur, acc[nn]);
        bcur = bnext;
      }
    }
    __syncthreads();
  }

  // ---- finalize and store O ----
#pragma unroll
  for (int r = 0; r < 8; ++r) {
    float inv = 1.f / l2[r];
#pragma unroll
    for (int nn = 0; nn < 8; ++nn) acc[nn][r] *= inv;
  }
#pragma unroll
  for (int nn = 0; nn < 8; ++nn) {
    int col = h * 128 + nn * 16 + (lane & 15);
#pragma unroll
    for (int r = 0; r < 8; ++r) {
      int rowg = i0 + wave * 16 + (lane >> 4) * 8 + r;
      O[(rowbase + rowg) * 1024 + col] = f2bf(acc[nn][r]);
    }
  }
}

// ---------------------------------------------------------------------------
// Host-side launch
// ---------------------------------------------------------------------------
extern "C" void kernel_launch(void* const* d_in, const int* in_sizes, int n_in,
                              void* d_out, int out_size, void* d_ws,
                              size_t ws_size, hipStream_t stream) {
  const float* s_v   = (const float*)d_in[0];
  const float* s_m   = (const float*)d_in[1];
  const float* g_l   = (const float*)d_in[2];
  const float* W_qkv = (const float*)d_in[3];
  const float* W_qk  = (const float*)d_in[4];
  const float* W_out = (const float*)d_in[5];
  const float* b_out = (const float*)d_in[6];
  float* out = (float*)d_out;

  const long NSV = 8L * 1024 * 1024;
  const long NWQKV = 1024L * 3072;
  const long NWQK = 1024L * 2048;
  const long NWOUT = 1024L * 1024;
  const long NQKVB = 8192L * 3072;
  const long NQKB = 8192L * 2048;
  const long NSTAT = 64L * 1024;
  const long NOBF = 8192L * 1024;
  const long NVT = 64L * 128 * 1024;   // vt[(b*8+h)*128 + d][j]

  char* ws = (char*)d_ws;
  size_t off = 0;
  auto carve = [&](size_t bytes) {
    void* p = ws + off;
    off += (bytes + 255) & ~(size_t)255;
    return p;
  };
  bf16* sv_bf   = (bf16*)carve(NSV * 2);
  bf16* sm_bf   = (bf16*)carve(NSV * 2);
  bf16* wqkvT   = (bf16*)carve(NWQKV * 2);   // [3072 x 1024]
  bf16* wqkT    = (bf16*)carve(NWQK * 2);    // [2048 x 1024]
  bf16* woutT   = (bf16*)carve(NWOUT * 2);   // [1024 x 1024]
  bf16* qkvb    = (bf16*)carve(NQKVB * 2);
  bf16* qkb     = (bf16*)carve(NQKB * 2);
  float* mstat  = (float*)carve(NSTAT * 4);
  float* lstat  = (float*)carve(NSTAT * 4);
  bf16* obf     = (bf16*)carve(NOBF * 2);
  bf16* vtb     = (bf16*)carve(NVT * 2);

  // 1) f32 -> bf16 conversions; weights converted + transposed to [N x K]
  cvt_f32_bf16<<<2048, 256, 0, stream>>>(s_v, sv_bf, NSV);
  cvt_f32_bf16<<<2048, 256, 0, stream>>>(s_m, sm_bf, NSV);
  cvt_transpose_bf16<<<dim3(3072 / 32, 1024 / 32), 256, 0, stream>>>(
      W_qkv, wqkvT, 1024, 3072);
  cvt_transpose_bf16<<<dim3(2048 / 32, 1024 / 32), 256, 0, stream>>>(
      W_qk, wqkT, 1024, 2048);
  cvt_transpose_bf16<<<dim3(1024 / 32, 1024 / 32), 256, 0, stream>>>(
      W_out, woutT, 1024, 1024);

  // 2) QKV / QK projection GEMMs
  gemm_bf16_wmma<<<dim3(3072 / 128, 8192 / 128), 256, 0, stream>>>(
      sv_bf, wqkvT, qkvb, nullptr, nullptr, 8192, 3072, 1024, 0);
  gemm_bf16_wmma<<<dim3(2048 / 128, 8192 / 128), 256, 0, stream>>>(
      sm_bf, wqkT, qkb, nullptr, nullptr, 8192, 2048, 1024, 0);

  // 3) transpose V slice for K-contiguous PV fragments
  transpose_v_bf16<<<dim3(1024 / 32, 128 / 32, 64), 256, 0, stream>>>(qkvb, vtb);

  // 4) softmax-m statistics
  attn_stats<<<512, 256, 0, stream>>>(qkb, mstat, lstat);

  // 5) fused flash attention (reads g_l exactly once)
  attn_flash<<<512, 256, 0, stream>>>(qkvb, qkb, vtb, g_l, mstat, lstat, obf);

  // 6) output projection + bias, f32 result
  gemm_bf16_wmma<<<dim3(1024 / 128, 8192 / 128), 256, 0, stream>>>(
      obf, woutT, nullptr, out, b_out, 8192, 1024, 1024, 1);
}
